// MAB_65068754534455
// MI455X (gfx1250) — matrix operand order repaired and checked by reading, not compiled, not run on hardware
//
#include <hip/hip_runtime.h>
#include <hip/hip_bf16.h>
#include <math.h>

// ---------- CDNA5 WMMA plumbing (wave32, v_wmma_f32_16x16x32_bf16) ----------
typedef __attribute__((ext_vector_type(16))) __bf16        v16bf;
typedef __attribute__((ext_vector_type(8)))  __bf16        v8bf;
typedef __attribute__((ext_vector_type(8)))  float         v8f;
typedef __attribute__((ext_vector_type(4)))  unsigned int  v4u;
typedef __attribute__((ext_vector_type(8)))  int           v8i;
typedef __attribute__((ext_vector_type(4)))  int           v4i;

__device__ __forceinline__ __bf16 tobf(float x) { return (__bf16)x; }

// Assemble a 16-elem bf16 fragment from two 16-byte LDS chunks (ds_load_b128 x2).
__device__ __forceinline__ v16bf ld16(const __bf16* p0, const __bf16* p1) {
  union U { v16bf v; v8bf h[2]; } u;
  u.h[0] = *(const v8bf*)p0;
  u.h[1] = *(const v8bf*)p1;
  return u.v;
}

__device__ __forceinline__ v8f wmma_bf16(v16bf a, v16bf b, v8f c) {
  return __builtin_amdgcn_wmma_f32_16x16x32_bf16(false, a, false, b, (short)0, c,
                                                 false, false);
}

// ---------------------------------------------------------------------------
// Tensor Data Mover: 2-D tile of bf16 (data_size=2B) global -> LDS.
// D# layout per CDNA5 ISA ch.8 (group0: flags/lds/global addr; group1: dims).
// width = contiguous dim-0 extent (elements), height = #rows,
// row_stride = elements between rows. Tight tensor dims (always in-bounds).
// Toolchain uses the 6-arg builtin: (v4u, v8i, v4i, v4i, v8i, i32 cpol).
// ---------------------------------------------------------------------------
__device__ __forceinline__ void tdm_load_2d(const void* lds_ptr, const void* gptr,
                                            unsigned int width, unsigned int height,
                                            unsigned int row_stride) {
  unsigned long long ga = (unsigned long long)(uintptr_t)gptr;
  unsigned int lds_off  = (unsigned int)(uintptr_t)lds_ptr;   // flat low 32 = LDS offset
  v4u g0;
  g0[0] = 1u;                                        // count=1, user descriptor
  g0[1] = lds_off;                                   // lds_addr (bytes)
  g0[2] = (unsigned int)ga;                          // global_addr[31:0]
  g0[3] = ((unsigned int)(ga >> 32) & 0x01FFFFFFu)   // global_addr[56:32]
          | 0x80000000u;                             // type=2 ("image")
  v8i g1;
  g1[0] = 0x00010000;                                // wg_mask=0, data_size=1 (2B)
  g1[1] = (int)((width & 0xffffu) << 16);            // tensor_dim0[15:0]
  g1[2] = (int)(((width >> 16) & 0xffffu) | ((height & 0xffffu) << 16));
  g1[3] = (int)(((height >> 16) & 0xffffu) | ((width & 0xffffu) << 16)); // tile_dim0
  g1[4] = (int)(height & 0xffffu);                   // tile_dim1, tile_dim2=0
  g1[5] = (int)row_stride;                           // tensor_dim0_stride[31:0]
  g1[6] = 0;                                         // stride0 hi / stride1 lo
  g1[7] = 0;
  v4i z4 = {0, 0, 0, 0};                             // groups 2/3 unused (2-D)
  v8i z8 = {0, 0, 0, 0, 0, 0, 0, 0};
  __builtin_amdgcn_tensor_load_to_lds(g0, g1, z4, z4, z8, 0);
}

// ---------------------------------------------------------------------------
// GEMM: Y[8192,512] = X[8192,512] @ W[512,512]^T + bias
//   Yf   : optional f32 output
//   Ybf  : optional bf16 output; bf_transposed=1 stores Ybf[col*8192+row]
//   relu_residual : Y = X + relu(X@W^T + b)   (FFN stage)
// Block tile 64(M) x 128(N), 8 waves; wave = 16(M) x 64(N) = 4 WMMA per K-step.
// ---------------------------------------------------------------------------
__global__ __launch_bounds__(256) void gemm_xwT_kernel(
    const float* __restrict__ X, const float* __restrict__ W,
    const float* __restrict__ bias, float* __restrict__ Yf,
    __bf16* __restrict__ Ybf, int bf_transposed, int relu_residual)
{
  __shared__ __attribute__((aligned(32))) __bf16 sX[64 * 32];
  __shared__ __attribute__((aligned(32))) __bf16 sW[128 * 32];

  const int tid  = threadIdx.x;
  const int lane = tid & 31;
  const int wave = tid >> 5;
  const int half = lane >> 4;
  const int l15  = lane & 15;

  const int bm = blockIdx.x * 64;
  const int bn = blockIdx.y * 128;
  const int wm = (wave & 3) * 16;
  const int wn = (wave >> 2) * 64;

  v8f acc[4] = {{}, {}, {}, {}};

  // staging maps: sX thread -> (row tid>>2, 8 cols); sW thread -> (row tid>>1, 16 cols)
  const int xr = tid >> 2, xc = (tid & 3) * 8;
  const int wr = tid >> 1, wc = (tid & 1) * 16;
  const float* gx = X + (size_t)(bm + xr) * 512 + xc;
  const float* gw = W + (size_t)(bn + wr) * 512 + wc;

  for (int k0 = 0; k0 < 512; k0 += 32) {
#pragma unroll
    for (int j = 0; j < 8; ++j)  sX[xr * 32 + xc + j] = tobf(gx[k0 + j]);
#pragma unroll
    for (int j = 0; j < 16; ++j) sW[wr * 32 + wc + j] = tobf(gw[k0 + j]);
    __syncthreads();
    if (k0 + 32 < 512) {                       // global_prefetch_b8 next K tiles
      __builtin_prefetch(gx + k0 + 32, 0, 1);
      __builtin_prefetch(gw + k0 + 32, 0, 1);
    }

    const __bf16* ax = &sX[(wm + l15) * 32 + half * 16];
    v16bf a = ld16(ax, ax + 8);
#pragma unroll
    for (int j = 0; j < 4; ++j) {
      const __bf16* bp = &sW[(wn + 16 * j + l15) * 32 + half * 16];
      acc[j] = wmma_bf16(a, ld16(bp, bp + 8), acc[j]);
    }
    __syncthreads();
  }

  const int rowbase = bm + wm + half * 8;
#pragma unroll
  for (int j = 0; j < 4; ++j) {
    const int col = bn + wn + 16 * j + l15;
    const float bia = bias[col];
#pragma unroll
    for (int r = 0; r < 8; ++r) {
      const size_t row = (size_t)(rowbase + r);
      float y = acc[j][r] + bia;
      if (relu_residual) y = X[row * 512 + col] + fmaxf(y, 0.f);
      if (Yf)  Yf[row * 512 + col] = y;
      if (Ybf) {
        if (bf_transposed) Ybf[(size_t)col * 8192 + row] = tobf(y);
        else               Ybf[row * 512 + col]          = tobf(y);
      }
    }
  }
}

// ---------------------------------------------------------------------------
// Flash attention per (b, h, 128-q-row block). 8 waves x 16 q rows each.
// bf16 Q/K/V^T tiles pulled into LDS by the Tensor Data Mover (1 slice/wave),
// scores + PV via WMMA, online softmax via 16-lane shuffles (C rows live in
// one half-wave). ctx = qf32 + softmax(q k^T / 8) v.
// ---------------------------------------------------------------------------
__global__ __launch_bounds__(256) void attn_kernel(
    const __bf16* __restrict__ Qbf, const __bf16* __restrict__ Kbf,
    const __bf16* __restrict__ Vt,  const float* __restrict__ Qf,
    float* __restrict__ ctx)
{
  __shared__ __attribute__((aligned(32))) __bf16 sQ [128 * 64]; // [qrow][hd]
  __shared__ __attribute__((aligned(32))) __bf16 sK [32 * 64];  // [key][hd]
  __shared__ __attribute__((aligned(32))) __bf16 sVt[64 * 32];  // [hd][key]
  __shared__ __attribute__((aligned(32))) __bf16 sP [8 * 16 * 32];

  const int tid  = threadIdx.x;
  const int lane = tid & 31;
  const int wave = tid >> 5;
  const int half = lane >> 4;
  const int l15  = lane & 15;

  const int b  = blockIdx.z;
  const int h  = blockIdx.y;
  const int q0 = blockIdx.x * 128;
  const size_t rowQ0 = (size_t)b * 2048 + q0;
  const size_t rowK0 = (size_t)b * 2048;
  const int hc = h * 64;

  // ---- TDM: each wave DMAs its 16-row slice of the Q tile (bf16, no cvt) ----
  const int qr = wave * 16;
  tdm_load_2d(&sQ[qr * 64], Qbf + (rowQ0 + qr) * 512 + hc, 64, 16, 512);
  __builtin_amdgcn_s_wait_tensorcnt(0);
  __syncthreads();

  v16bf aq[2];
#pragma unroll
  for (int s = 0; s < 2; ++s) {
    const __bf16* p = &sQ[(qr + l15) * 64 + 32 * s + half * 16];
    aq[s] = ld16(p, p + 8);
  }

  v8f o[4] = {{}, {}, {}, {}};
  v8f mrun, lrun;
#pragma unroll
  for (int r = 0; r < 8; ++r) { mrun[r] = -INFINITY; lrun[r] = 0.f; }

  __bf16* pb = &sP[wave * 16 * 32];

  for (int c0 = 0; c0 < 2048; c0 += 32) {
    // ---- TDM: wave slices of K chunk [32x64] and V^T chunk [64x32] ----
    tdm_load_2d(&sK[wave * 4 * 64],
                Kbf + (rowK0 + c0 + wave * 4) * 512 + hc, 64, 4, 512);
    tdm_load_2d(&sVt[wave * 8 * 32],
                Vt + (size_t)(hc + wave * 8) * 8192 + rowK0 + c0, 32, 8, 8192);
    __builtin_amdgcn_s_wait_tensorcnt(0);
    __syncthreads();

    // ---- scores: S = Q(16x64) @ K^T(64x32) -> two 16x16 tiles ----
    v8f s0 = {}, s1 = {};
#pragma unroll
    for (int s = 0; s < 2; ++s) {
      const __bf16* bp0 = &sK[(l15)      * 64 + 32 * s + half * 16];
      const __bf16* bp1 = &sK[(16 + l15) * 64 + 32 * s + half * 16];
      s0 = wmma_bf16(aq[s], ld16(bp0, bp0 + 8), s0);
      s1 = wmma_bf16(aq[s], ld16(bp1, bp1 + 8), s1);
    }

    // ---- online softmax; each C row lives in one 16-lane half ----
#pragma unroll
    for (int r = 0; r < 8; ++r) {
      float a0 = s0[r] * 0.125f;
      float a1 = s1[r] * 0.125f;
      float mloc = fmaxf(a0, a1);
#pragma unroll
      for (int off = 1; off < 16; off <<= 1)
        mloc = fmaxf(mloc, __shfl_xor(mloc, off));
      const float mnew = fmaxf(mrun[r], mloc);
      const float fac  = __expf(mrun[r] - mnew);
      const float p0   = __expf(a0 - mnew);
      const float p1   = __expf(a1 - mnew);
      float rs = p0 + p1;
#pragma unroll
      for (int off = 1; off < 16; off <<= 1)
        rs += __shfl_xor(rs, off);
      lrun[r] = lrun[r] * fac + rs;
      mrun[r] = mnew;
      o[0][r] *= fac; o[1][r] *= fac; o[2][r] *= fac; o[3][r] *= fac;

      const int ml = r + half * 8;                  // C-layout -> [m][key]
      pb[ml * 32 + l15]      = tobf(p0);
      pb[ml * 32 + 16 + l15] = tobf(p1);
    }

    // ---- O += P(16x32) @ V(32x64) ----
    const __bf16* ap0 = &pb[l15 * 32 + half * 16];
    v16bf ap = ld16(ap0, ap0 + 8);
#pragma unroll
    for (int j = 0; j < 4; ++j) {
      const __bf16* bv = &sVt[(16 * j + l15) * 32 + half * 16];
      o[j] = wmma_bf16(ap, ld16(bv, bv + 8), o[j]);
    }
    __syncthreads();
  }

  // ---- finalize: ctx = qf32 + O / l ----
  v8f inv;
#pragma unroll
  for (int r = 0; r < 8; ++r) inv[r] = 1.0f / lrun[r];
  const size_t outrow0 = rowQ0 + qr + half * 8;
#pragma unroll
  for (int j = 0; j < 4; ++j) {
    const int col = hc + 16 * j + l15;
#pragma unroll
    for (int r = 0; r < 8; ++r) {
      const size_t row = outrow0 + r;
      ctx[row * 512 + col] = Qf[row * 512 + col] + o[j][r] * inv[r];
    }
  }
}

// ---------------------------------------------------------------------------
extern "C" void kernel_launch(void* const* d_in, const int* in_sizes, int n_in,
                              void* d_out, int out_size, void* d_ws, size_t ws_size,
                              hipStream_t stream) {
  (void)in_sizes; (void)n_in; (void)out_size; (void)ws_size;
  const float* query = (const float*)d_in[0];
  const float* keyv  = (const float*)d_in[1];
  const float* Wq = (const float*)d_in[2];
  const float* bq = (const float*)d_in[3];
  const float* Wk = (const float*)d_in[4];
  const float* bk = (const float*)d_in[5];
  const float* Wv = (const float*)d_in[6];
  const float* bv = (const float*)d_in[7];
  const float* Wo = (const float*)d_in[8];
  const float* bo = (const float*)d_in[9];
  float* out = (float*)d_out;

  const size_t MN = (size_t)4 * 2048 * 512;
  float*  qf   = (float*)d_ws;               // f32 q (residual)        16 MB
  float*  ctx  = qf + MN;                    // f32 attention output    16 MB
  __bf16* qbf  = (__bf16*)(ctx + MN);        // bf16 q                   8 MB
  __bf16* kbf  = qbf + MN;                   // bf16 k                   8 MB
  __bf16* vtbf = kbf + MN;                   // bf16 v, [col][row]       8 MB

  dim3 gp(8192 / 64, 512 / 128);             // 128 x 4 blocks
  dim3 tb(256);

  gemm_xwT_kernel<<<gp, tb, 0, stream>>>(query, Wq, bq, qf,      qbf,  0, 0);
  gemm_xwT_kernel<<<gp, tb, 0, stream>>>(keyv,  Wk, bk, nullptr, kbf,  0, 0);
  gemm_xwT_kernel<<<gp, tb, 0, stream>>>(keyv,  Wv, bv, nullptr, vtbf, 1, 0);

  dim3 ga(2048 / 128, 8, 4);                 // (q-blocks, heads, batch)
  attn_kernel<<<ga, tb, 0, stream>>>(qbf, kbf, vtbf, qf, ctx);

  gemm_xwT_kernel<<<gp, tb, 0, stream>>>(ctx, Wo, bo, out, nullptr, 0, 1);
}
